// ArcFaceLoss_65068754534622
// MI455X (gfx1250) — matrix operand order repaired and checked by reading, not compiled, not run on hardware
//
#include <hip/hip_runtime.h>
#include <math.h>

typedef __attribute__((ext_vector_type(16))) _Float16 v16h;
typedef __attribute__((ext_vector_type(8)))  _Float16 v8h;
typedef __attribute__((ext_vector_type(8)))  float    v8f;

constexpr int   EMB_DIM = 512;
constexpr int   BATCH   = 2048;
constexpr int   CLASSES = 50000;
constexpr float SCALEF  = 64.0f;
constexpr float COS_M   = 0.8253356149096783f;   // cos(0.6)
constexpr float SIN_M   = 0.5646424733950354f;   // sin(0.6)
constexpr float TH_C    = -0.8253356149096783f;  // cos(pi-0.6)
constexpr float MM_C    = 0.33878548403702124f;  // sin(pi-0.6)*0.6
constexpr float EPS_C   = 1e-7f;

// ws layout (bytes)
constexpr size_t WS_EMBN = 0;                                   // 2 MB f16
constexpr size_t WS_RNW  = 2u * 1024u * 1024u;                  // 200 KB f32
constexpr size_t WS_NLL  = 2u * 1024u * 1024u + 256u * 1024u;   // 8 KB f32
constexpr size_t WS_W16  = 4u * 1024u * 1024u;                  // 51.2 MB f16 (optional)
constexpr size_t WS_BIG  = WS_W16 + (size_t)CLASSES * EMB_DIM * 2 + 1024;

// ---------------------------------------------------------------------------
// L2-normalize emb rows -> f16 (wave per row)
// ---------------------------------------------------------------------------
__global__ void __launch_bounds__(256) normalize_emb_k(const float* __restrict__ emb,
                                                       _Float16* __restrict__ embn) {
    int wave = (blockIdx.x * blockDim.x + threadIdx.x) >> 5;
    int lane = threadIdx.x & 31;
    if (wave >= BATCH) return;
    const float* row = emb + (size_t)wave * EMB_DIM;
    float v[16];
    float ss = 0.f;
#pragma unroll
    for (int i = 0; i < 16; ++i) { v[i] = row[lane + 32 * i]; ss += v[i] * v[i]; }
#pragma unroll
    for (int off = 16; off; off >>= 1) ss += __shfl_xor(ss, off, 32);
    float rn = 1.0f / fmaxf(sqrtf(ss), 1e-12f);
    _Float16* orow = embn + (size_t)wave * EMB_DIM;
#pragma unroll
    for (int i = 0; i < 16; ++i) orow[lane + 32 * i] = (_Float16)(v[i] * rn);
}

// ---------------------------------------------------------------------------
// per-row 1/||W_row|| only (fallback path)
// ---------------------------------------------------------------------------
__global__ void __launch_bounds__(256) wnorm_k(const float* __restrict__ W,
                                               float* __restrict__ rnw) {
    int wave = (blockIdx.x * blockDim.x + threadIdx.x) >> 5;
    int lane = threadIdx.x & 31;
    if (wave >= CLASSES) return;
    const float* row = W + (size_t)wave * EMB_DIM;
    float ss = 0.f;
#pragma unroll
    for (int i = 0; i < 16; ++i) { float x = row[lane + 32 * i]; ss += x * x; }
#pragma unroll
    for (int off = 16; off; off >>= 1) ss += __shfl_xor(ss, off, 32);
    if (lane == 0) rnw[wave] = 1.0f / fmaxf(sqrtf(ss), 1e-12f);
}

// ---------------------------------------------------------------------------
// normalize W rows -> f16 W_n (big-workspace path; rnw folded in)
// ---------------------------------------------------------------------------
__global__ void __launch_bounds__(256) wnorm16_k(const float* __restrict__ W,
                                                 _Float16* __restrict__ W16) {
    int wave = (blockIdx.x * blockDim.x + threadIdx.x) >> 5;
    int lane = threadIdx.x & 31;
    if (wave >= CLASSES) return;
    const float* row = W + (size_t)wave * EMB_DIM;
    float v[16];
    float ss = 0.f;
#pragma unroll
    for (int i = 0; i < 16; ++i) { v[i] = row[lane + 32 * i]; ss += v[i] * v[i]; }
#pragma unroll
    for (int off = 16; off; off >>= 1) ss += __shfl_xor(ss, off, 32);
    float rn = 1.0f / fmaxf(sqrtf(ss), 1e-12f);
    _Float16* orow = W16 + (size_t)wave * EMB_DIM;
#pragma unroll
    for (int i = 0; i < 16; ++i) orow[lane + 32 * i] = (_Float16)(v[i] * rn);
}

// ---------------------------------------------------------------------------
// Shared fragment helpers
// ---------------------------------------------------------------------------
__device__ __forceinline__ v16h load_a_frag(const _Float16* __restrict__ p) {
    // A 16x32 layout: 16 contiguous K per lane group (matches B)
    v8h lo = *(const v8h*)(p);
    v8h hi = *(const v8h*)(p + 8);
    v16h a;
#pragma unroll
    for (int i = 0; i < 8; ++i) { a[i] = lo[i]; a[i + 8] = hi[i]; }
    return a;
}

__device__ __forceinline__ v16h load_b_frag16(const _Float16* __restrict__ p) {
    // Dense B 32x16 layout: 16 contiguous K per lane group
    v8h lo = *(const v8h*)(p);
    v8h hi = *(const v8h*)(p + 8);
    v16h b;
#pragma unroll
    for (int i = 0; i < 8; ++i) { b[i] = lo[i]; b[i + 8] = hi[i]; }
    return b;
}

__device__ __forceinline__ v16h load_b_frag32(const float* __restrict__ p) {
    v8f lo = *(const v8f*)(p);
    v8f hi = *(const v8f*)(p + 8);
    v16h b;
#pragma unroll
    for (int i = 0; i < 8; ++i) {
        b[i]     = (_Float16)lo[i];
        b[i + 8] = (_Float16)hi[i];
    }
    return b;
}

__device__ __forceinline__ float clip_scale(float cosv) {
    cosv = fminf(fmaxf(cosv, -1.0f + EPS_C), 1.0f - EPS_C);
    return cosv * SCALEF;
}

// ---------------------------------------------------------------------------
// GEMM, big-ws path: pre-normalized f16 W. Wave = 32M x 64N (8 WMMA/K-step),
// block = 8 waves = 32M x 512N.
// ---------------------------------------------------------------------------
__global__ void __launch_bounds__(256, 2) gemm16_k(const _Float16* __restrict__ embn,
                                                   const _Float16* __restrict__ W16,
                                                   float* __restrict__ logits) {
    const int lane    = threadIdx.x & 31;
    const int wave    = threadIdx.x >> 5;
    const int mBase   = blockIdx.y * 32;
    const int nBase   = blockIdx.x * 512 + wave * 64;
    const int halfSel = lane >> 4;
    const int lr      = lane & 15;

    const _Float16* aRow0 = embn + (size_t)(mBase + lr) * EMB_DIM + halfSel * 16;
    const _Float16* aRow1 = aRow0 + (size_t)16 * EMB_DIM;

    int             ncol[4];
    const _Float16* bRow[4];
#pragma unroll
    for (int t = 0; t < 4; ++t) {
        int n   = nBase + t * 16 + lr;
        ncol[t] = n;
        int nc  = n < CLASSES ? n : (CLASSES - 1);
        bRow[t] = W16 + (size_t)nc * EMB_DIM + halfSel * 16;
    }

    v8f c0[4] = {v8f{}, v8f{}, v8f{}, v8f{}};
    v8f c1[4] = {v8f{}, v8f{}, v8f{}, v8f{}};

    for (int k = 0; k < EMB_DIM; k += 32) {
        v16h a0 = load_a_frag(aRow0 + k);
        v16h a1 = load_a_frag(aRow1 + k);
#pragma unroll
        for (int t = 0; t < 4; ++t) {
            v16h b = load_b_frag16(bRow[t] + k);
            c0[t] = __builtin_amdgcn_wmma_f32_16x16x32_f16(
                false, a0, false, b, (short)0, c0[t], false, false);
            c1[t] = __builtin_amdgcn_wmma_f32_16x16x32_f16(
                false, a1, false, b, (short)0, c1[t], false, false);
        }
    }

    const int rowOff0 = mBase + halfSel * 8;
#pragma unroll
    for (int t = 0; t < 4; ++t) {
        if (ncol[t] < CLASSES) {
#pragma unroll
            for (int r = 0; r < 8; ++r) {
                __builtin_nontemporal_store(
                    clip_scale(c0[t][r]),
                    logits + (size_t)(rowOff0 + r) * CLASSES + ncol[t]);
                __builtin_nontemporal_store(
                    clip_scale(c1[t][r]),
                    logits + (size_t)(rowOff0 + 16 + r) * CLASSES + ncol[t]);
            }
        }
    }
}

// ---------------------------------------------------------------------------
// GEMM, fallback path: f32 W converted in-register, rnw applied post-loop.
// Same 32M x 64N wave tile.
// ---------------------------------------------------------------------------
__global__ void __launch_bounds__(256, 2) gemm32_k(const _Float16* __restrict__ embn,
                                                   const float* __restrict__ W,
                                                   const float* __restrict__ rnw,
                                                   float* __restrict__ logits) {
    const int lane    = threadIdx.x & 31;
    const int wave    = threadIdx.x >> 5;
    const int mBase   = blockIdx.y * 32;
    const int nBase   = blockIdx.x * 512 + wave * 64;
    const int halfSel = lane >> 4;
    const int lr      = lane & 15;

    const _Float16* aRow0 = embn + (size_t)(mBase + lr) * EMB_DIM + halfSel * 16;
    const _Float16* aRow1 = aRow0 + (size_t)16 * EMB_DIM;

    int          ncol[4];
    const float* bRow[4];
    float        rn[4];
#pragma unroll
    for (int t = 0; t < 4; ++t) {
        int n   = nBase + t * 16 + lr;
        ncol[t] = n;
        int nc  = n < CLASSES ? n : (CLASSES - 1);
        bRow[t] = W + (size_t)nc * EMB_DIM + halfSel * 16;
        rn[t]   = rnw[nc];
    }

    v8f c0[4] = {v8f{}, v8f{}, v8f{}, v8f{}};
    v8f c1[4] = {v8f{}, v8f{}, v8f{}, v8f{}};

    for (int k = 0; k < EMB_DIM; k += 32) {
        v16h a0 = load_a_frag(aRow0 + k);
        v16h a1 = load_a_frag(aRow1 + k);
#pragma unroll
        for (int t = 0; t < 4; ++t) {
            v16h b = load_b_frag32(bRow[t] + k);
            c0[t] = __builtin_amdgcn_wmma_f32_16x16x32_f16(
                false, a0, false, b, (short)0, c0[t], false, false);
            c1[t] = __builtin_amdgcn_wmma_f32_16x16x32_f16(
                false, a1, false, b, (short)0, c1[t], false, false);
        }
    }

    const int rowOff0 = mBase + halfSel * 8;
#pragma unroll
    for (int t = 0; t < 4; ++t) {
        if (ncol[t] < CLASSES) {
            float rnt = rn[t];
#pragma unroll
            for (int r = 0; r < 8; ++r) {
                __builtin_nontemporal_store(
                    clip_scale(c0[t][r] * rnt),
                    logits + (size_t)(rowOff0 + r) * CLASSES + ncol[t]);
                __builtin_nontemporal_store(
                    clip_scale(c1[t][r] * rnt),
                    logits + (size_t)(rowOff0 + 16 + r) * CLASSES + ncol[t]);
            }
        }
    }
}

// ---------------------------------------------------------------------------
// per-row online softmax + NLL with margin substitution at label
// ---------------------------------------------------------------------------
__global__ void __launch_bounds__(256) row_softmax_nll_k(const float* __restrict__ logits,
                                                         const int* __restrict__ label,
                                                         float* __restrict__ nll) {
    const int row = blockIdx.x;
    const int tid = threadIdx.x;
    const int lab = label[row];
    const float* lrow = logits + (size_t)row * CLASSES;

    float cosT = lrow[lab] * (1.0f / SCALEF);
    float sine = fmaxf(sqrtf(fmaxf(1.0f - cosT * cosT, 0.0f)), EPS_C);
    float phi  = cosT * COS_M - sine * SIN_M;
    float tl   = ((cosT > TH_C) ? phi : (cosT - MM_C)) * SCALEF;

    float m = -3.0e38f, s = 0.0f;
    for (int j = tid; j < CLASSES; j += 256) {
        float x = __builtin_nontemporal_load(lrow + j);
        if (j == lab) x = tl;
        float mn = fmaxf(m, x);
        s = s * __expf(m - mn) + __expf(x - mn);
        m = mn;
    }
#pragma unroll
    for (int off = 16; off; off >>= 1) {
        float m2 = __shfl_xor(m, off, 32);
        float s2 = __shfl_xor(s, off, 32);
        float mn = fmaxf(m, m2);
        s = s * __expf(m - mn) + s2 * __expf(m2 - mn);
        m = mn;
    }
    __shared__ float smax[8], ssum[8];
    if ((tid & 31) == 0) { smax[tid >> 5] = m; ssum[tid >> 5] = s; }
    __syncthreads();
    if (tid == 0) {
        float M = smax[0], S = ssum[0];
#pragma unroll
        for (int w = 1; w < 8; ++w) {
            float mn = fmaxf(M, smax[w]);
            S = S * __expf(M - mn) + ssum[w] * __expf(smax[w] - mn);
            M = mn;
        }
        nll[row] = M + logf(S) - tl;
    }
}

// ---------------------------------------------------------------------------
// mean of 2048 NLLs -> out[0]
// ---------------------------------------------------------------------------
__global__ void __launch_bounds__(256) final_mean_k(const float* __restrict__ nll,
                                                    float* __restrict__ out) {
    __shared__ float buf[8];
    int tid = threadIdx.x;
    float s = 0.f;
    for (int i = tid; i < BATCH; i += 256) s += nll[i];
#pragma unroll
    for (int off = 16; off; off >>= 1) s += __shfl_xor(s, off, 32);
    if ((tid & 31) == 0) buf[tid >> 5] = s;
    __syncthreads();
    if (tid == 0) {
        float t = 0.f;
#pragma unroll
        for (int w = 0; w < 8; ++w) t += buf[w];
        out[0] = t * (1.0f / (float)BATCH);
    }
}

// ---------------------------------------------------------------------------
extern "C" void kernel_launch(void* const* d_in, const int* in_sizes, int n_in,
                              void* d_out, int out_size, void* d_ws, size_t ws_size,
                              hipStream_t stream) {
    const float* emb   = (const float*)d_in[0];
    const int*   label = (const int*)d_in[1];
    const float* W     = (const float*)d_in[2];
    float*       out   = (float*)d_out;

    char*     ws   = (char*)d_ws;
    _Float16* embn = (_Float16*)(ws + WS_EMBN);
    float*    rnw  = (float*)(ws + WS_RNW);
    float*    nll  = (float*)(ws + WS_NLL);
    _Float16* W16  = (_Float16*)(ws + WS_W16);

    float* logits = out + 1;   // d_out = [loss, logits(2048x50000)]

    normalize_emb_k<<<BATCH / 8, 256, 0, stream>>>(emb, embn);

    dim3 g3((CLASSES + 511) / 512, BATCH / 32);
    if (ws_size >= WS_BIG) {
        wnorm16_k<<<(CLASSES + 7) / 8, 256, 0, stream>>>(W, W16);
        gemm16_k<<<g3, 256, 0, stream>>>(embn, W16, logits);
    } else {
        wnorm_k<<<(CLASSES + 7) / 8, 256, 0, stream>>>(W, rnw);
        gemm32_k<<<g3, 256, 0, stream>>>(embn, W, rnw, logits);
    }

    row_softmax_nll_k<<<BATCH, 256, 0, stream>>>(logits, label, nll);
    final_mean_k<<<1, 256, 0, stream>>>(nll, out);
}
